// LegalGAT_60653528154443
// MI455X (gfx1250) — matrix operand order repaired and checked by reading, not compiled, not run on hardware
//
#include <hip/hip_runtime.h>
#include <cstddef>

#define DEVINL __device__ __forceinline__

typedef __attribute__((ext_vector_type(16))) __bf16 v16bf;
typedef __attribute__((ext_vector_type(8)))  float  v8f;

struct alignas(16) UQ { unsigned int x, y, z, w; };
struct UO { UQ lo, hi; };

DEVINL unsigned short f2bf(float f) {
  unsigned int u = __float_as_uint(f);
  unsigned int r = u + 0x7FFFu + ((u >> 16) & 1u);   // RNE
  return (unsigned short)(r >> 16);
}

DEVINL float lrelu(float x) { return x > 0.f ? x : 0.2f * x; }

// ---------------------------------------------------------------------------
// f32 -> bf16 elementwise convert
// ---------------------------------------------------------------------------
__global__ void f32_to_bf16_kernel(const float* __restrict__ in,
                                   unsigned short* __restrict__ out, int n) {
  int i = blockIdx.x * blockDim.x + threadIdx.x;
  if (i < n) out[i] = f2bf(in[i]);
}

// ---------------------------------------------------------------------------
// Pack W (K x Ncols row-major f32) into per-lane WMMA B layout (bf16):
//   element idx -> j=idx&15, lane=(idx>>4)&31, t=idx>>9, nt=t%NT, ks=t/NT
//   source K = ks*32 + 8*(lane>=16) + (j<8 ? j : j+8) ; col = nt*16 + (lane&15)
// ---------------------------------------------------------------------------
__global__ void pack_B_kernel(const float* __restrict__ W,
                              unsigned short* __restrict__ Bp,
                              int K, int Ncols) {
  int idx = blockIdx.x * blockDim.x + threadIdx.x;
  int total = K * Ncols;
  if (idx >= total) return;
  int NT   = Ncols >> 4;
  int j    = idx & 15;
  int lane = (idx >> 4) & 31;
  int t    = idx >> 9;
  int nt   = t % NT;
  int ks   = t / NT;
  int kk   = ks * 32 + ((lane >= 16) ? 8 : 0) + (j < 8 ? j : j + 8);
  int col  = nt * 16 + (lane & 15);
  Bp[idx]  = f2bf(W[(size_t)kk * Ncols + col]);
}

// ---------------------------------------------------------------------------
// WMMA bf16 GEMM: C[M x NCOLS] = A[M x KSTEPS*32] * Bpack
// Block = 128 threads (4 waves). Wave w owns n-tiles [w*NTPW, (w+1)*NTPW);
// each wave computes MTPW M-tiles so every B-tile is reused MTPW times.
// Grid.x = M / (16*MTPW).
// ---------------------------------------------------------------------------
template <int KSTEPS, int NCOLS, int NTPW, int MTPW>
__global__ void wmma_gemm_bf16(const unsigned short* __restrict__ A,
                               const unsigned short* __restrict__ Bp,
                               float* __restrict__ C, int M) {
  const int lane = threadIdx.x & 31;
  const int wave = threadIdx.x >> 5;
  const int m0   = blockIdx.x * (16 * MTPW);
  const int bsel = (lane >> 4) & 1;            // half-wave select
  const int K    = KSTEPS * 32;
  const int NT   = NCOLS >> 4;
  const int nt0  = wave * NTPW;

  // A per-lane base: row (lane&15), K chunks [bsel*8, +8) and [bsel*8+16, +8)
  const unsigned short* arow[MTPW];
#pragma unroll
  for (int mt = 0; mt < MTPW; ++mt)
    arow[mt] = A + (size_t)(m0 + mt * 16 + (lane & 15)) * K + bsel * 8;

  v8f acc[MTPW][NTPW];
#pragma unroll
  for (int mt = 0; mt < MTPW; ++mt)
#pragma unroll
    for (int i = 0; i < NTPW; ++i) acc[mt][i] = (v8f){};

#pragma unroll 2
  for (int ks = 0; ks < KSTEPS; ++ks) {
    v16bf a[MTPW];
#pragma unroll
    for (int mt = 0; mt < MTPW; ++mt) {
      UO ua;
      ua.lo = *reinterpret_cast<const UQ*>(arow[mt] + ks * 32);
      ua.hi = *reinterpret_cast<const UQ*>(arow[mt] + ks * 32 + 16);
      a[mt] = __builtin_bit_cast(v16bf, ua);
      if (ks + 2 < KSTEPS)                     // stream A ahead of use
        __builtin_prefetch(arow[mt] + (ks + 2) * 32, 0, 1);
    }
#pragma unroll
    for (int i = 0; i < NTPW; ++i) {
      const unsigned short* bp =
          Bp + (((size_t)ks * NT + (nt0 + i)) * 32 + lane) * 16;
      UO ub;
      ub.lo = *reinterpret_cast<const UQ*>(bp);
      ub.hi = *reinterpret_cast<const UQ*>(bp + 8);
      v16bf b = __builtin_bit_cast(v16bf, ub);
#pragma unroll
      for (int mt = 0; mt < MTPW; ++mt)
        acc[mt][i] = __builtin_amdgcn_wmma_f32_16x16x32_bf16(
            false, a[mt], false, b, (short)0, acc[mt][i], false, false);
    }
  }

  const int col0 = lane & 15;
#pragma unroll
  for (int mt = 0; mt < MTPW; ++mt) {
    const int rbase = m0 + mt * 16 + bsel * 8;
#pragma unroll
    for (int i = 0; i < NTPW; ++i) {
      const int cb = (nt0 + i) * 16 + col0;
#pragma unroll
      for (int r = 0; r < 8; ++r)
        C[(size_t)(rbase + r) * NCOLS + cb] = acc[mt][i][r];
    }
  }
}

// ---------------------------------------------------------------------------
// Per-node attention score precompute: as[n,h] = <feat[n,h,:], a_src[h,:]>
// ---------------------------------------------------------------------------
template <int H>
__global__ void node_scores(const float* __restrict__ feat,
                            const float* __restrict__ a_src,
                            const float* __restrict__ a_dst,
                            float* __restrict__ as, float* __restrict__ ad,
                            int n) {
  int i = blockIdx.x * blockDim.x + threadIdx.x;
  if (i >= n * H) return;
  int node = i / H, h = i % H;
  const float* fp = feat + (size_t)node * (H * 64) + h * 64;
  const float* sp = a_src + h * 64;
  const float* dp = a_dst + h * 64;
  float ss = 0.f, dd = 0.f;
#pragma unroll 8
  for (int c = 0; c < 64; ++c) {
    float v = fp[c];
    ss = fmaf(v, sp[c], ss);
    dd = fmaf(v, dp[c], dd);
  }
  as[i] = ss;
  ad[i] = dd;
}

// ---------------------------------------------------------------------------
// CSR build: count, single-block scan, fill
// ---------------------------------------------------------------------------
__global__ void zero_i32(int* __restrict__ p, int n) {
  int i = blockIdx.x * blockDim.x + threadIdx.x;
  if (i < n) p[i] = 0;
}

__global__ void count_deg(const long long* __restrict__ ei, int E, int N,
                          int* __restrict__ deg) {
  int e = blockIdx.x * blockDim.x + threadIdx.x;
  if (e < E) {
    int d = (int)ei[(size_t)E + e];   // row 1 = dst
    if (d >= 0 && d < N) atomicAdd(&deg[d], 1);
  }
}

__global__ void scan_kernel(const int* __restrict__ deg,
                            int* __restrict__ rowstart, int n) {
  __shared__ int sm[1024];
  __shared__ int sbase;
  if (threadIdx.x == 0) sbase = 0;
  __syncthreads();
  for (int start = 0; start < n; start += 1024) {
    int i = start + (int)threadIdx.x;
    int v = (i < n) ? deg[i] : 0;
    sm[threadIdx.x] = v;
    __syncthreads();
    for (int off = 1; off < 1024; off <<= 1) {
      int t = (threadIdx.x >= (unsigned)off) ? sm[threadIdx.x - off] : 0;
      __syncthreads();
      sm[threadIdx.x] += t;
      __syncthreads();
    }
    int incl = sm[threadIdx.x];
    if (i < n) rowstart[i] = sbase + incl - v;   // exclusive prefix
    __syncthreads();
    if (threadIdx.x == 1023) sbase += sm[1023];
    __syncthreads();
  }
  if (threadIdx.x == 0) rowstart[n] = sbase;
}

__global__ void fill_csr(const long long* __restrict__ ei, int E, int N,
                         const int* __restrict__ rowstart,
                         int* __restrict__ cursor, int* __restrict__ csr_src) {
  int e = blockIdx.x * blockDim.x + threadIdx.x;
  if (e < E) {
    int d = (int)ei[(size_t)E + e];
    int s = (int)ei[e];
    if (d >= 0 && d < N) {
      int pos = atomicAdd(&cursor[d], 1);
      csr_src[rowstart[d] + pos] = s;
    }
  }
}

// ---------------------------------------------------------------------------
// Segment softmax weight precompute: one thread per (node, head).
// Writes normalized self-loop weight and per-CSR-slot edge weights, so the
// aggregation pass is a pure gather-FMA (no redundant exp/lrelu per wave).
// ---------------------------------------------------------------------------
template <int H>
__global__ void softmax_weights(const float* __restrict__ as,
                                const float* __restrict__ ad,
                                const int* __restrict__ rowstart,
                                const int* __restrict__ csr_src,
                                float* __restrict__ wself,   // N*H
                                float* __restrict__ wcsr,    // E*H (CSR order)
                                int N) {
  int i = blockIdx.x * blockDim.x + threadIdx.x;
  if (i >= N * H) return;
  int n = i / H, h = i % H;
  const float adv = ad[n * H + h];
  const float es  = lrelu(as[n * H + h] + adv);   // self-loop score
  const int rs = rowstart[n], re = rowstart[n + 1];

  float m = es;
  for (int k = rs; k < re; ++k)
    m = fmaxf(m, lrelu(as[csr_src[k] * H + h] + adv));

  float pself = __expf(es - m);
  float denom = pself;
  for (int k = rs; k < re; ++k) {
    float p = __expf(lrelu(as[csr_src[k] * H + h] + adv) - m);
    wcsr[(size_t)k * H + h] = p;
    denom += p;
  }
  const float inv = 1.0f / fmaxf(denom, 1e-16f);
  wself[i] = pself * inv;
  for (int k = rs; k < re; ++k) wcsr[(size_t)k * H + h] *= inv;
}

// ---------------------------------------------------------------------------
// Aggregation: one block per node, H*64 threads; pure weighted gather.
// Optionally fuses +bias, ELU, and bf16 emit for the next GEMM.
// ---------------------------------------------------------------------------
template <int H, bool APPLY_ELU>
__global__ void gat_aggregate(const float* __restrict__ feat,    // N x H*64
                              const float* __restrict__ wself,   // N x H
                              const float* __restrict__ wcsr,    // E x H
                              const int* __restrict__ rowstart,
                              const int* __restrict__ csr_src,
                              const float* __restrict__ bias,    // H*64
                              float* __restrict__ out_f32,
                              unsigned short* __restrict__ out_bf16,
                              int N) {
  const int n   = blockIdx.x;
  const int tid = threadIdx.x;          // 0 .. H*64-1
  const int h   = tid >> 6;
  const int rs = rowstart[n], re = rowstart[n + 1];

  float acc = wself[n * H + h] * feat[(size_t)n * (H * 64) + tid];
  for (int k = rs; k < re; ++k) {
    int s   = csr_src[k];                       // wave-uniform
    float w = wcsr[(size_t)k * H + h];          // wave-uniform broadcast
    acc = fmaf(w, feat[(size_t)s * (H * 64) + tid], acc);  // coalesced
  }
  float v = acc + bias[tid];
  if (APPLY_ELU) v = (v > 0.f) ? v : (__expf(v) - 1.0f);
  if (out_f32)  out_f32[(size_t)n * (H * 64) + tid] = v;
  if (out_bf16) out_bf16[(size_t)n * (H * 64) + tid] = f2bf(v);
}

// ---------------------------------------------------------------------------
// Host-side launcher
// ---------------------------------------------------------------------------
static inline char* ws_take(char*& p, size_t bytes) {
  char* r = p;
  p += (bytes + 255) & ~(size_t)255;
  return r;
}

extern "C" void kernel_launch(void* const* d_in, const int* in_sizes, int n_in,
                              void* d_out, int out_size, void* d_ws,
                              size_t ws_size, hipStream_t stream) {
  (void)in_sizes; (void)n_in; (void)out_size; (void)ws_size;
  constexpr int N = 20000, E = 320000, FIN = 512, HID = 64, HEADS = 4;
  constexpr int C1 = HEADS * HID;   // 256
  constexpr int C2 = HID;           // 64

  const float*     x      = (const float*)d_in[0];
  const long long* ei     = (const long long*)d_in[1];
  const float*     W1     = (const float*)d_in[2];
  const float*     a_src1 = (const float*)d_in[3];
  const float*     a_dst1 = (const float*)d_in[4];
  const float*     b1     = (const float*)d_in[5];
  const float*     W2     = (const float*)d_in[6];
  const float*     a_src2 = (const float*)d_in[7];
  const float*     a_dst2 = (const float*)d_in[8];
  const float*     b2     = (const float*)d_in[9];
  float* out = (float*)d_out;

  char* w = (char*)d_ws;
  unsigned short* xbf  = (unsigned short*)ws_take(w, (size_t)N * FIN * 2);
  unsigned short* w1p  = (unsigned short*)ws_take(w, (size_t)FIN * C1 * 2);
  float*          h1   = (float*)ws_take(w, (size_t)N * C1 * 4);
  float*          as1  = (float*)ws_take(w, (size_t)N * HEADS * 4);
  float*          ad1  = (float*)ws_take(w, (size_t)N * HEADS * 4);
  int*            deg  = (int*)ws_take(w, (size_t)N * 4);
  int*            cur  = (int*)ws_take(w, (size_t)N * 4);
  int*            rst  = (int*)ws_take(w, (size_t)(N + 1) * 4);
  int*            csr  = (int*)ws_take(w, (size_t)E * 4);
  float*          wsf  = (float*)ws_take(w, (size_t)N * HEADS * 4);
  float*          wcs  = (float*)ws_take(w, (size_t)E * HEADS * 4);
  unsigned short* helu = (unsigned short*)ws_take(w, (size_t)N * C1 * 2);
  unsigned short* w2p  = (unsigned short*)ws_take(w, (size_t)C1 * C2 * 2);
  float*          h2   = (float*)ws_take(w, (size_t)N * C2 * 4);
  float*          as2  = (float*)ws_take(w, (size_t)N * 4);
  float*          ad2  = (float*)ws_take(w, (size_t)N * 4);

  const int TB = 256;
  // --- layer 1 GEMM: h1 = x @ W1 (bf16 WMMA, fp32 acc) ---
  f32_to_bf16_kernel<<<(N * FIN + TB - 1) / TB, TB, 0, stream>>>(x, xbf, N * FIN);
  pack_B_kernel<<<(FIN * C1 + TB - 1) / TB, TB, 0, stream>>>(W1, w1p, FIN, C1);
  wmma_gemm_bf16<FIN / 32, C1, 4, 2><<<N / 32, 128, 0, stream>>>(xbf, w1p, h1, N);

  // --- per-node scores for layer 1 ---
  node_scores<HEADS><<<(N * HEADS + TB - 1) / TB, TB, 0, stream>>>(
      h1, a_src1, a_dst1, as1, ad1, N);

  // --- CSR over real edges (self-loops handled in softmax/aggregation) ---
  zero_i32<<<(N + TB - 1) / TB, TB, 0, stream>>>(deg, N);
  zero_i32<<<(N + TB - 1) / TB, TB, 0, stream>>>(cur, N);
  count_deg<<<(E + TB - 1) / TB, TB, 0, stream>>>(ei, E, N, deg);
  scan_kernel<<<1, 1024, 0, stream>>>(deg, rst, N);
  fill_csr<<<(E + TB - 1) / TB, TB, 0, stream>>>(ei, E, N, rst, cur, csr);

  // --- layer 1 segment softmax + aggregation (+b1, ELU, bf16 emit) ---
  softmax_weights<HEADS><<<(N * HEADS + TB - 1) / TB, TB, 0, stream>>>(
      as1, ad1, rst, csr, wsf, wcs, N);
  gat_aggregate<HEADS, true><<<N, C1, 0, stream>>>(
      h1, wsf, wcs, rst, csr, b1, nullptr, helu, N);

  // --- layer 2 GEMM: h2 = helu @ W2 ---
  pack_B_kernel<<<(C1 * C2 + TB - 1) / TB, TB, 0, stream>>>(W2, w2p, C1, C2);
  wmma_gemm_bf16<C1 / 32, C2, 1, 2><<<N / 32, 128, 0, stream>>>(helu, w2p, h2, N);

  // --- layer 2 scores + softmax + aggregation (+b2) -> out ---
  node_scores<1><<<(N + TB - 1) / TB, TB, 0, stream>>>(h2, a_src2, a_dst2,
                                                       as2, ad2, N);
  softmax_weights<1><<<(N + TB - 1) / TB, TB, 0, stream>>>(
      as2, ad2, rst, csr, wsf, wcs, N);
  gat_aggregate<1, false><<<N, C2, 0, stream>>>(
      h2, wsf, wcs, rst, csr, b2, out, nullptr, N);
}
